// HierarchicalVQ_26551487824081
// MI455X (gfx1250) — matrix operand order repaired and checked by reading, not compiled, not run on hardware
//
#include <hip/hip_runtime.h>
#include <hip/hip_bf16.h>

#define B_    128
#define N_    256
#define D_    256
#define KP_   1024
#define KS_   512
#define MP_   (B_ * N_)       // 32768 patch rows
#define BETA_ 0.25f
#define EPS_  1e-5f

typedef __attribute__((ext_vector_type(16))) __bf16 v16bf;
typedef __attribute__((ext_vector_type(8)))  __bf16 v8bf;
typedef __attribute__((ext_vector_type(8)))  float  v8f;

// ---------------------------------------------------------------------------
// A-fragment (16x32 bf16, MxK) from f32 source: lane L holds row M=L%16.
// Per ISA: VGPR0..3 => K = {0..7}+8*half ; VGPR4..7 => K = {16..23}+8*half
// => element e: K = (e%8) + 8*half + 16*(e/8): two contiguous runs of 8.
// Converted once per wave, outside the hot loop.
// ---------------------------------------------------------------------------
__device__ __forceinline__ v16bf load_fragA_f32(const float* __restrict__ p, int k0) {
    float4 f0 = *(const float4*)(p + k0);
    float4 f1 = *(const float4*)(p + k0 + 4);
    float4 f2 = *(const float4*)(p + k0 + 16);
    float4 f3 = *(const float4*)(p + k0 + 20);
    v16bf a;
    a[0]  = (__bf16)f0.x; a[1]  = (__bf16)f0.y; a[2]  = (__bf16)f0.z; a[3]  = (__bf16)f0.w;
    a[4]  = (__bf16)f1.x; a[5]  = (__bf16)f1.y; a[6]  = (__bf16)f1.z; a[7]  = (__bf16)f1.w;
    a[8]  = (__bf16)f2.x; a[9]  = (__bf16)f2.y; a[10] = (__bf16)f2.z; a[11] = (__bf16)f2.w;
    a[12] = (__bf16)f3.x; a[13] = (__bf16)f3.y; a[14] = (__bf16)f3.z; a[15] = (__bf16)f3.w;
    return a;
}

// ---------------------------------------------------------------------------
// B-fragment (32x16 bf16, KxN) from pre-converted bf16 codebook: lane L holds
// column N=L%16; element e: K = (e%8) + 8*half + 16*(e/8) (matches A).
// Two b128 loads straight into the WMMA source registers — no conversion.
// ---------------------------------------------------------------------------
__device__ __forceinline__ v16bf load_fragB_bf(const __bf16* __restrict__ p, int k0) {
    v8bf lo = *(const v8bf*)(p + k0);
    v8bf hi = *(const v8bf*)(p + k0 + 16);
    return __builtin_shufflevector(lo, hi,
                                   0, 1, 2, 3, 4, 5, 6, 7,
                                   8, 9, 10, 11, 12, 13, 14, 15);
}

// ---------------------------------------------------------------------------
// f32 -> bf16 conversion (grid-stride)
// ---------------------------------------------------------------------------
__global__ void cvt_bf16(const float* __restrict__ in, __bf16* __restrict__ out, int n) {
    for (int i = blockIdx.x * blockDim.x + threadIdx.x; i < n; i += gridDim.x * blockDim.x)
        out[i] = (__bf16)in[i];
}

// ---------------------------------------------------------------------------
// WMMA-based VQ argmin: one wave per 16-row tile of X. Sweeps K codes in
// 16-wide column tiles; 8 chained v_wmma_f32_16x16x32_bf16 per tile (D=256).
// dist(m,n) = ||c_n||^2 - 2*score (||x||^2 is row-constant, irrelevant).
// ---------------------------------------------------------------------------
__global__ void __launch_bounds__(256, 1)
vq_argmin_wmma(const float* __restrict__ X,
               const __bf16* __restrict__ CB,   // pre-converted bf16 codebook
               const float* __restrict__ CN,
               int M, int K,
               int* __restrict__ outIdx) {
    const int lane = threadIdx.x & 31;
    const int wave = threadIdx.x >> 5;
    const int rowTile = blockIdx.x * (blockDim.x >> 5) + wave;
    if (rowTile * 16 >= M) return;          // wave-uniform: EXEC stays all-1s

    const int half = lane >> 4;
    const int r    = lane & 15;

    // Stage all A fragments for this wave's 16 rows (full D=256)
    const float* xrow = X + (size_t)(rowTile * 16 + r) * D_;
    v16bf a[8];
#pragma unroll
    for (int kc = 0; kc < 8; ++kc)
        a[kc] = load_fragA_f32(xrow, kc * 32 + 8 * half);

    float bestVal[8];
    int   bestIdx[8];
#pragma unroll
    for (int v = 0; v < 8; ++v) { bestVal[v] = 3.4e38f; bestIdx[v] = 0; }

    const int nTiles = K >> 4;
    for (int ct = 0; ct < nTiles; ++ct) {
        const __bf16* crow = CB + (size_t)(ct * 16 + r) * D_;
        v8f acc = {};
#pragma unroll
        for (int kc = 0; kc < 8; ++kc) {
            v16bf b = load_fragB_bf(crow, kc * 32 + 8 * half);
            acc = __builtin_amdgcn_wmma_f32_16x16x32_bf16(
                false, a[kc], false, b, (short)0, acc, false, false);
        }
        const int   n  = ct * 16 + r;       // this lane's column (code id)
        const float cn = CN[n];
#pragma unroll
        for (int v = 0; v < 8; ++v) {
            float dist = cn - 2.0f * acc[v];
            if (dist < bestVal[v]) { bestVal[v] = dist; bestIdx[v] = n; }
        }
    }

    // Cross-lane argmin within each 16-lane half (xor<=8 stays in half).
    // C/D layout: slot v <-> row (v + 8*half) of the tile, N = lane%16.
#pragma unroll
    for (int v = 0; v < 8; ++v) {
        float val = bestVal[v];
        int   idx = bestIdx[v];
#pragma unroll
        for (int off = 8; off >= 1; off >>= 1) {
            float ov = __shfl_xor(val, off, 32);
            int   oi = __shfl_xor(idx, off, 32);
            if (ov < val || (ov == val && oi < idx)) { val = ov; idx = oi; }
        }
        if (r == 0)
            outIdx[rowTile * 16 + half * 8 + v] = idx;
    }
}

// ---------------------------------------------------------------------------
// Per-row squared norms of a codebook (one row per block, D_=256 threads)
// ---------------------------------------------------------------------------
__global__ void row_sqnorm(const float* __restrict__ A, float* __restrict__ out) {
    __shared__ float s[D_];
    float v = A[(size_t)blockIdx.x * D_ + threadIdx.x];
    s[threadIdx.x] = v * v;
    __syncthreads();
    for (int st = 128; st > 0; st >>= 1) {
        if ((int)threadIdx.x < st) s[threadIdx.x] += s[threadIdx.x + st];
        __syncthreads();
    }
    if (threadIdx.x == 0) out[blockIdx.x] = s[0];
}

// ---------------------------------------------------------------------------
// Gather q = CB[idx], write quantized output + float index, accumulate
// sum((q-x)^2) into *acc, optionally histogram into patch_dist.
// One row per block, 256 threads (one per D column).
// ---------------------------------------------------------------------------
__global__ void gather_loss(const float* __restrict__ X,
                            const float* __restrict__ CB,
                            const int* __restrict__ idxArr,
                            float* __restrict__ outQ,
                            float* __restrict__ outIdxF,
                            float* __restrict__ pdist,   // nullptr => no hist
                            float* __restrict__ acc) {
    __shared__ float s[D_];
    const int row = blockIdx.x;
    const int idx = idxArr[row];
    const float x = X[(size_t)row * D_ + threadIdx.x];
    const float q = CB[(size_t)idx * D_ + threadIdx.x];
    outQ[(size_t)row * D_ + threadIdx.x] = q;
    const float d = q - x;
    s[threadIdx.x] = d * d;
    __syncthreads();
    for (int st = 128; st > 0; st >>= 1) {
        if ((int)threadIdx.x < st) s[threadIdx.x] += s[threadIdx.x + st];
        __syncthreads();
    }
    if (threadIdx.x == 0) {
        atomicAdd(acc, s[0]);
        outIdxF[row] = (float)idx;
        if (pdist) atomicAdd(&pdist[(size_t)(row / N_) * KP_ + idx], 1.0f / (float)N_);
    }
}

// ---------------------------------------------------------------------------
// y = LN(in @ W + b; g, beta) [optionally ReLU]. One batch row per block,
// 256 threads = one output column each; Kdim is the reduction depth.
// ---------------------------------------------------------------------------
__global__ void mlp_ln(const float* __restrict__ in,
                       const float* __restrict__ W,
                       const float* __restrict__ bb,
                       const float* __restrict__ g,
                       const float* __restrict__ beta,
                       float* __restrict__ out,
                       int Kdim, int doRelu) {
    __shared__ float s[D_];
    const int brow = blockIdx.x;
    const int col  = threadIdx.x;
    const float* x = in + (size_t)brow * Kdim;
    float y = bb[col];
    for (int k = 0; k < Kdim; ++k)
        y = fmaf(x[k], W[(size_t)k * D_ + col], y);
    // mean
    s[col] = y; __syncthreads();
    for (int st = 128; st > 0; st >>= 1) {
        if (col < st) s[col] += s[col + st];
        __syncthreads();
    }
    const float mu = s[0] * (1.0f / (float)D_);
    __syncthreads();
    const float c = y - mu;
    s[col] = c * c; __syncthreads();
    for (int st = 128; st > 0; st >>= 1) {
        if (col < st) s[col] += s[col + st];
        __syncthreads();
    }
    const float var = s[0] * (1.0f / (float)D_);
    float o = c * rsqrtf(var + EPS_) * g[col] + beta[col];
    if (doRelu) o = fmaxf(o, 0.0f);
    out[(size_t)brow * D_ + col] = o;
}

__global__ void finalize_loss(const float* __restrict__ acc, float* __restrict__ out) {
    out[0] = (1.0f + BETA_) *
             (acc[0] / (float)(MP_ * D_) + acc[1] / (float)(B_ * D_));
}

// ---------------------------------------------------------------------------
extern "C" void kernel_launch(void* const* d_in, const int* in_sizes, int n_in,
                              void* d_out, int out_size, void* d_ws, size_t ws_size,
                              hipStream_t stream) {
    (void)in_sizes; (void)n_in; (void)out_size; (void)ws_size;

    const float* pe    = (const float*)d_in[0];   // [B,N,D]
    const float* pcb   = (const float*)d_in[1];   // [KP,D]
    const float* scb   = (const float*)d_in[2];   // [KS,D]
    const float* W1    = (const float*)d_in[3];   // [KP,D]
    const float* b1    = (const float*)d_in[4];
    const float* g1    = (const float*)d_in[5];
    const float* beta1 = (const float*)d_in[6];
    const float* W2    = (const float*)d_in[7];   // [D,D]
    const float* b2    = (const float*)d_in[8];
    const float* g2    = (const float*)d_in[9];
    const float* beta2 = (const float*)d_in[10];

    float* out = (float*)d_out;
    float* ws  = (float*)d_ws;

    // workspace layout (float elements; bf16 regions 16B-aligned)
    float*  acc   = ws;                            // [2] loss accumulators
    float*  cnp   = ws + 4;                        // [KP] patch code norms
    float*  cns   = cnp + KP_;                     // [KS] state code norms
    int*    pidx  = (int*)(cns + KS_);             // [MP] patch indices
    int*    sidx  = pidx + MP_;                    // [B]  state indices
    float*  pdist = (float*)(sidx + B_);           // [B,KP] histogram
    float*  h     = pdist + (size_t)B_ * KP_;      // [B,D]
    float*  semb  = h + (size_t)B_ * D_;           // [B,D]
    __bf16* pcbbf = (__bf16*)(semb + (size_t)B_ * D_);       // [KP,D] bf16
    __bf16* scbbf = pcbbf + (size_t)KP_ * D_;                // [KS,D] bf16

    // d_out layout (concatenated, float)
    float* outQ    = out;                         // [B,N,D]
    float* outSQ   = outQ + (size_t)MP_ * D_;     // [B,D]
    float* outLoss = outSQ + (size_t)B_ * D_;     // [1]
    float* outPIdx = outLoss + 1;                 // [B,N]
    float* outSIdx = outPIdx + (size_t)MP_;       // [B]

    hipMemsetAsync(acc, 0, 2 * sizeof(float), stream);
    hipMemsetAsync(pdist, 0, (size_t)B_ * KP_ * sizeof(float), stream);

    row_sqnorm<<<KP_, D_, 0, stream>>>(pcb, cnp);
    row_sqnorm<<<KS_, D_, 0, stream>>>(scb, cns);
    cvt_bf16<<<256, 256, 0, stream>>>(pcb, pcbbf, KP_ * D_);
    cvt_bf16<<<128, 256, 0, stream>>>(scb, scbbf, KS_ * D_);

    // Patch VQ: 2048 row-tiles, 8 waves/block -> 256 blocks
    vq_argmin_wmma<<<MP_ / (16 * 8), 256, 0, stream>>>(pe, pcbbf, cnp, MP_, KP_, pidx);
    gather_loss<<<MP_, D_, 0, stream>>>(pe, pcb, pidx, outQ, outPIdx, pdist, acc);

    mlp_ln<<<B_, D_, 0, stream>>>(pdist, W1, b1, g1, beta1, h, KP_, 1);
    mlp_ln<<<B_, D_, 0, stream>>>(h, W2, b2, g2, beta2, semb, D_, 0);

    // State VQ: 8 row-tiles -> one block of 8 waves
    vq_argmin_wmma<<<1, 256, 0, stream>>>(semb, scbbf, cns, B_, KS_, sidx);
    gather_loss<<<B_, D_, 0, stream>>>(semb, scb, sidx, outSQ, outSIdx, nullptr, acc + 1);

    finalize_loss<<<1, 1, 0, stream>>>(acc, outLoss);
}
